// MP_up_14001593385536
// MI455X (gfx1250) — matrix operand order repaired and checked by reading, not compiled, not run on hardware
//
#include <hip/hip_runtime.h>
#include <stdint.h>

#define DD 128
#define HH 8

typedef __attribute__((ext_vector_type(16))) __bf16 v16bf;
typedef __attribute__((ext_vector_type(8)))  float  v8f;

__device__ __forceinline__ unsigned short f2bf(float f) {
  union { float f; unsigned int u; } c; c.f = f;
  unsigned int u = c.u;
  unsigned int r = u + 0x7FFFu + ((u >> 16) & 1u);
  return (unsigned short)(r >> 16);
}

// ---------------- elementwise f32 -> bf16 ----------------
__global__ void k_cvt_bf16(const float* __restrict__ x, unsigned short* __restrict__ y, int n4) {
  int i = blockIdx.x * blockDim.x + threadIdx.x;
  if (i >= n4) return;
  float4 v = ((const float4*)x)[i];
  ushort4 o;
  o.x = f2bf(v.x); o.y = f2bf(v.y); o.z = f2bf(v.z); o.w = f2bf(v.w);
  ((ushort4*)y)[i] = o;
}

// ---------------- zero fill ----------------
__global__ void k_zero(float* __restrict__ p, int n4) {
  int i = blockIdx.x * blockDim.x + threadIdx.x;
  if (i >= n4) return;
  ((float4*)p)[i] = make_float4(0.f, 0.f, 0.f, 0.f);
}

// ---------------- pack row-major W[K,Nd] (f32) into WMMA B-fragment stream (bf16) --------
// fragment id = ((ntile*(K/32) + kb)*32 + lane); each lane holds 16 contiguous bf16:
//   n = ntile*16 + (lane&15);  k = kb*32 + (lane>>4)*16 + t, t=0..15
__global__ void k_pack_w(const float* __restrict__ W, unsigned short* __restrict__ out,
                         int Kdim, int Nd) {
  int fid = blockIdx.x * blockDim.x + threadIdx.x;
  int kblocks = Kdim >> 5;
  int total = (Nd >> 4) * kblocks * 32;
  if (fid >= total) return;
  int lane  = fid & 31;
  int kb    = (fid >> 5) % kblocks;
  int ntile = fid / (kblocks << 5);
  int n     = (ntile << 4) + (lane & 15);
  int kbase = (kb << 5) + ((lane >> 4) << 4);
  unsigned short frag[16];
#pragma unroll
  for (int t = 0; t < 16; ++t) frag[t] = f2bf(W[(kbase + t) * Nd + n]);
  uint4* dst = (uint4*)(out + fid * 16);
  dst[0] = *(uint4*)(frag);
  dst[1] = *(uint4*)(frag + 8);
}

// ---------------- WMMA GEMM:  C[M,NDIM] = A[M,KDIM](bf16) @ Bpacked + bias  --------------
// one block = one 16-row tile, one wave per 16-col tile (NDIM/16 waves).
template<int KDIM, int NDIM, bool RELU, bool OUTF32, bool OUTBF16>
__global__ void k_gemm_wmma(const unsigned short* __restrict__ A,
                            const unsigned short* __restrict__ Bp,
                            const float* __restrict__ bias,
                            float* __restrict__ Cf,
                            unsigned short* __restrict__ Cb,
                            int M) {
  __shared__ unsigned short As[16 * KDIM];
  const int tid  = threadIdx.x;
  const int lane = tid & 31;
  const int wave = tid >> 5;
  const int half = lane >> 4;
  const int r    = lane & 15;
  const int rowbase = blockIdx.x << 4;
  const int nthreads = (NDIM / 16) * 32;

  // stage 16 x KDIM A tile into LDS (b128 per thread), int addressing only
#pragma unroll
  for (int idx = tid * 8; idx < 16 * KDIM; idx += nthreads * 8) {
    int ar = idx / KDIM;
    int ac = idx - ar * KDIM;
    int grow = rowbase + ar; if (grow >= M) grow = M - 1;
    *(uint4*)(As + idx) = *(const uint4*)(A + grow * KDIM + ac);
  }
  __syncthreads();

  v8f acc = {};
  const int ntile = wave;
#pragma unroll
  for (int kb = 0; kb < KDIM / 32; ++kb) {
    union { uint4 u[2]; v16bf v; } a, b;
    // A fragment: lanes 0-15 rows, K runs [base, base+16) contiguous, base = half*16
    const unsigned short* ap = As + r * KDIM + kb * 32 + half * 16;
    a.u[0] = *(const uint4*)(ap);
    a.u[1] = *(const uint4*)(ap + 8);
    // B fragment: pre-packed, 16 contiguous bf16 per lane
    const unsigned short* bp = Bp + (((ntile * (KDIM / 32) + kb) << 5) + lane) * 16;
    b.u[0] = *(const uint4*)(bp);
    b.u[1] = *(const uint4*)(bp + 8);
    acc = __builtin_amdgcn_wmma_f32_16x16x32_bf16(false, a.v, false, b.v,
                                                  (short)0, acc, false, false);
  }

  const int col = (ntile << 4) + r;
  const float bvv = bias[col];
  const int mybase = rowbase + (half << 3);        // first of my 8 rows
  if (rowbase + 16 <= M) {
    // fast path: unconditional, constant-stride stores
#pragma unroll
    for (int v = 0; v < 8; ++v) {
      float val = acc[v] + bvv;
      if (RELU) val = fmaxf(val, 0.0f);
      int off = (mybase + v) * NDIM + col;
      if (OUTF32)  Cf[off] = val;
      if (OUTBF16) Cb[off] = f2bf(val);
    }
  } else {
#pragma unroll
    for (int v = 0; v < 8; ++v) {
      float val = acc[v] + bvv;
      if (RELU) val = fmaxf(val, 0.0f);
      int rowm = mybase + v;
      if (rowm < M) {
        int off = rowm * NDIM + col;
        if (OUTF32)  Cf[off] = val;
        if (OUTBF16) Cb[off] = f2bf(val);
      }
    }
  }
}

// ---------------- edge attention: one wave per edge ----------------
__global__ void k_edge_attn(const int* __restrict__ src, const int* __restrict__ dst,
                            const float* __restrict__ Kt, const float* __restrict__ Qt,
                            const float* __restrict__ Vt,
                            float* __restrict__ wV, float* __restrict__ z, int E) {
  const int lane = threadIdx.x & 31;
  const int wavesPerBlock = blockDim.x >> 5;
  int wid = blockIdx.x * wavesPerBlock + (threadIdx.x >> 5);
  const int nW = gridDim.x * wavesPerBlock;
  const int c4 = lane * 4;
  for (int e = wid; e < E; e += nW) {
    int s = src[e], d = dst[e];
    int en = e + nW;
    if (en < E) {               // prefetch next iteration's rows (global_prefetch_b8)
      int sn = src[en], dn = dst[en];
      __builtin_prefetch(Kt + sn * DD + c4, 0, 3);
      __builtin_prefetch(Qt + dn * DD + c4, 0, 3);
      __builtin_prefetch(Vt + sn * DD + c4, 0, 3);
    }
    float4 kv = *(const float4*)(Kt + s * DD + c4);
    float4 qv = *(const float4*)(Qt + d * DD + c4);
    float p = kv.x * qv.x + kv.y * qv.y + kv.z * qv.z + kv.w * qv.w;
    // per-head (16 dims = 4 lanes) reduction
    p += __shfl_xor(p, 1, 32);
    p += __shfl_xor(p, 2, 32);
    float sc = __expf(fminf(fmaxf(p * 0.25f, -5.0f), 5.0f));   // 1/sqrt(16)=0.25
    float4 vv = *(const float4*)(Vt + s * DD + c4);
    float* wp = wV + d * DD + c4;
    unsafeAtomicAdd(wp + 0, vv.x * sc);
    unsafeAtomicAdd(wp + 1, vv.y * sc);
    unsafeAtomicAdd(wp + 2, vv.z * sc);
    unsafeAtomicAdd(wp + 3, vv.w * sc);
    if ((lane & 3) == 0) unsafeAtomicAdd(z + d * HH + (lane >> 2), sc);
  }
}

// ---------------- attn = wV / (z + 1e-6)  -> bf16 ----------------
__global__ void k_attn_div(const float* __restrict__ wV, const float* __restrict__ z,
                           unsigned short* __restrict__ outb, int N) {
  int i = blockIdx.x * blockDim.x + threadIdx.x;   // one float4 per thread
  int total = N * 32;
  if (i >= total) return;
  int row = i >> 5;
  int g   = i & 31;                                 // cols 4g..4g+3, head = g>>2
  float inv = 1.0f / (z[row * HH + (g >> 2)] + 1e-6f);
  float4 v = ((const float4*)wV)[i];
  ushort4 o;
  o.x = f2bf(v.x * inv); o.y = f2bf(v.y * inv);
  o.z = f2bf(v.z * inv); o.w = f2bf(v.w * inv);
  ((ushort4*)outb)[i] = o;
}

// ---------------- out = LN(base + add)*g + b ; emit f32 + bf16 ----------------
__global__ void k_residual_ln(const float* __restrict__ base, const float* __restrict__ add,
                              const float* __restrict__ g, const float* __restrict__ b,
                              float* __restrict__ outf, unsigned short* __restrict__ outb,
                              int N) {
  const int lane = threadIdx.x & 31;
  int row = blockIdx.x * (blockDim.x >> 5) + (threadIdx.x >> 5);
  if (row >= N) return;
  int off = row * DD + lane * 4;
  float4 x = *(const float4*)(base + off);
  float4 a = *(const float4*)(add + off);
  x.x += a.x; x.y += a.y; x.z += a.z; x.w += a.w;
  float s = x.x + x.y + x.z + x.w;
#pragma unroll
  for (int o = 1; o < 32; o <<= 1) s += __shfl_xor(s, o, 32);
  float mean = s * (1.0f / DD);
  float dx = x.x - mean, dy = x.y - mean, dz = x.z - mean, dw = x.w - mean;
  float vs = dx * dx + dy * dy + dz * dz + dw * dw;
#pragma unroll
  for (int o = 1; o < 32; o <<= 1) vs += __shfl_xor(vs, o, 32);
  float rstd = rsqrtf(vs * (1.0f / DD) + 1e-5f);
  float4 gg = *(const float4*)(g + lane * 4);
  float4 bb = *(const float4*)(b + lane * 4);
  float4 o4;
  o4.x = dx * rstd * gg.x + bb.x;
  o4.y = dy * rstd * gg.y + bb.y;
  o4.z = dz * rstd * gg.z + bb.z;
  o4.w = dw * rstd * gg.w + bb.w;
  *(float4*)(outf + off) = o4;
  ushort4 ob;
  ob.x = f2bf(o4.x); ob.y = f2bf(o4.y); ob.z = f2bf(o4.z); ob.w = f2bf(o4.w);
  *(ushort4*)(outb + off) = ob;
}

extern "C" void kernel_launch(void* const* d_in, const int* in_sizes, int n_in,
                              void* d_out, int out_size, void* d_ws, size_t ws_size,
                              hipStream_t stream) {
  const float* h0  = (const float*)d_in[0];
  const int*   src = (const int*)d_in[1];
  const int*   dst = (const int*)d_in[2];
  const float* Wq = (const float*)d_in[3];  const float* bq = (const float*)d_in[4];
  const float* Wk = (const float*)d_in[5];  const float* bk = (const float*)d_in[6];
  const float* Wv = (const float*)d_in[7];  const float* bv = (const float*)d_in[8];
  const float* Wo = (const float*)d_in[9];  const float* bo = (const float*)d_in[10];
  const float* W1 = (const float*)d_in[11]; const float* b1 = (const float*)d_in[12];
  const float* W2 = (const float*)d_in[13]; const float* b2 = (const float*)d_in[14];
  const float* g1  = (const float*)d_in[15]; const float* be1 = (const float*)d_in[16];
  const float* g2  = (const float*)d_in[17]; const float* be2 = (const float*)d_in[18];

  const int N = in_sizes[0] / DD;
  const int E = in_sizes[1];
  const int L = 2;

  // ---- workspace layout (256B aligned) ----
  char* wsp = (char*)d_ws;
  auto alloc = [&](size_t bytes) -> void* {
    void* p = (void*)wsp;
    wsp += (bytes + 255) & ~(size_t)255;
    return p;
  };
  unsigned short* hb  = (unsigned short*)alloc((size_t)N * DD * 2); // bf16 A operand (reused)
  float* Qb   = (float*)alloc((size_t)N * DD * 4);                  // Q   / attn-out
  float* Kb   = (float*)alloc((size_t)N * DD * 4);                  // K   / post-LN1 f32
  float* Vb   = (float*)alloc((size_t)N * DD * 4);                  // V   / ffn1 bf16 [N,256]
  float* wVb  = (float*)alloc((size_t)N * DD * 4);                  // wV  / ffn2 out
  float* zb   = (float*)alloc((size_t)N * HH * 4);
  float* hcur = (float*)alloc((size_t)N * DD * 4);                  // layer output f32
  unsigned short *pWq[2], *pWk[2], *pWv[2], *pWo[2], *pW1[2], *pW2[2];
  for (int l = 0; l < L; ++l) {
    pWq[l] = (unsigned short*)alloc((size_t)DD * DD * 2);
    pWk[l] = (unsigned short*)alloc((size_t)DD * DD * 2);
    pWv[l] = (unsigned short*)alloc((size_t)DD * DD * 2);
    pWo[l] = (unsigned short*)alloc((size_t)DD * DD * 2);
    pW1[l] = (unsigned short*)alloc((size_t)DD * 2 * DD * 2);
    pW2[l] = (unsigned short*)alloc((size_t)2 * DD * DD * 2);
  }

  auto pack = [&](const float* W, unsigned short* out, int Kd, int Nd) {
    int total = (Nd / 16) * (Kd / 32) * 32;
    k_pack_w<<<(total + 255) / 256, 256, 0, stream>>>(W, out, Kd, Nd);
  };
  for (int l = 0; l < L; ++l) {
    pack(Wq + (size_t)l * DD * DD, pWq[l], DD, DD);
    pack(Wk + (size_t)l * DD * DD, pWk[l], DD, DD);
    pack(Wv + (size_t)l * DD * DD, pWv[l], DD, DD);
    pack(Wo + (size_t)l * DD * DD, pWo[l], DD, DD);
    pack(W1 + (size_t)l * DD * 2 * DD, pW1[l], DD, 2 * DD);
    pack(W2 + (size_t)l * 2 * DD * DD, pW2[l], 2 * DD, DD);
  }

  {
    int n4 = N * DD / 4;
    k_cvt_bf16<<<(n4 + 255) / 256, 256, 0, stream>>>(h0, hb, n4);
  }

  const int mtiles = (N + 15) / 16;
  const float* cur = h0;
  for (int l = 0; l < L; ++l) {
    // Q, K, V projections
    k_gemm_wmma<DD, DD, false, true, false><<<mtiles, 256, 0, stream>>>(hb, pWq[l], bq + (size_t)l * DD, Qb, nullptr, N);
    k_gemm_wmma<DD, DD, false, true, false><<<mtiles, 256, 0, stream>>>(hb, pWk[l], bk + (size_t)l * DD, Kb, nullptr, N);
    k_gemm_wmma<DD, DD, false, true, false><<<mtiles, 256, 0, stream>>>(hb, pWv[l], bv + (size_t)l * DD, Vb, nullptr, N);
    // clear accumulators
    {
      int n4 = N * DD / 4;
      k_zero<<<(n4 + 255) / 256, 256, 0, stream>>>(wVb, n4);
      int z4 = N * HH / 4;
      k_zero<<<(z4 + 255) / 256, 256, 0, stream>>>(zb, z4);
    }
    // edge phase (L2-resident gathers + f32 atomics)
    k_edge_attn<<<2048, 256, 0, stream>>>(src, dst, Kb, Qb, Vb, wVb, zb, E);
    // attn normalize -> bf16 (into hb)
    {
      int t = N * 32;
      k_attn_div<<<(t + 255) / 256, 256, 0, stream>>>(wVb, zb, hb, N);
    }
    // output projection -> Qb
    k_gemm_wmma<DD, DD, false, true, false><<<mtiles, 256, 0, stream>>>(hb, pWo[l], bo + (size_t)l * DD, Qb, nullptr, N);
    // residual + LN1 -> Kb (f32) + hb (bf16)
    k_residual_ln<<<(N + 7) / 8, 256, 0, stream>>>(cur, Qb, g1 + (size_t)l * DD, be1 + (size_t)l * DD, Kb, hb, N);
    // FFN up (ReLU, bf16 out into Vb region)
    k_gemm_wmma<DD, 2 * DD, true, false, true><<<mtiles, 512, 0, stream>>>(hb, pW1[l], b1 + (size_t)l * 2 * DD, nullptr, (unsigned short*)Vb, N);
    // FFN down -> wVb
    k_gemm_wmma<2 * DD, DD, false, true, false><<<mtiles, 256, 0, stream>>>((const unsigned short*)Vb, pW2[l], b2 + (size_t)l * DD, wVb, nullptr, N);
    // residual + LN2 -> layer output (d_out on final layer) + hb bf16 for next layer
    float* outf = (l == L - 1) ? (float*)d_out : hcur;
    k_residual_ln<<<(N + 7) / 8, 256, 0, stream>>>(Kb, wVb, g2 + (size_t)l * DD, be2 + (size_t)l * DD, outf, hb, N);
    cur = hcur;
  }
}
